// GNN_1_interaction_logp_46024869544461
// MI455X (gfx1250) — matrix operand order repaired and checked by reading, not compiled, not run on hardware
//
#include <hip/hip_runtime.h>
#include <hip/hip_bf16.h>
#include <math.h>

typedef __attribute__((ext_vector_type(16))) _Float16 v16h;
typedef __attribute__((ext_vector_type(8)))  float    v8f;

#define NSOL 6144
#define NWAT 6144
#define NOCT 6144
#define ESOL 24576
#define EOCT 24576
#define NBATCH 192
#define EMBD 128

#ifndef __has_builtin
#define __has_builtin(x) 0
#endif
#if __has_builtin(__builtin_amdgcn_global_load_async_to_lds_b128)
#define ASYNC_LDS 1
#else
#define ASYNC_LDS 0
#endif

#if ASYNC_LDS
// async copy of 16B global -> LDS, tracked by ASYNCcnt (CDNA5 GLOBAL_LOAD_ASYNC_TO_LDS_B128)
// builtin params (per hipcc diagnostic): (int4 AS1* src, int4 AS3* dst, Imm offset, Imm cpol)
typedef int ai4 __attribute__((vector_size(16)));
typedef __attribute__((address_space(1))) ai4* as1_i4p;
typedef __attribute__((address_space(3))) ai4* as3_i4p;
#define ASYNC_CP16(dst, src)                                             \
  __builtin_amdgcn_global_load_async_to_lds_b128(                        \
      (as1_i4p)(const void*)(src), (as3_i4p)(void*)(dst), 0, 0)
__device__ __forceinline__ void async_wait0() {
#if __has_builtin(__builtin_amdgcn_s_wait_asynccnt)
  __builtin_amdgcn_s_wait_asynccnt(0);
#else
  asm volatile("s_wait_asynccnt 0" ::: "memory");
#endif
}
#endif

// ---------------------------------------------------------------------------
// WMMA fragment loaders (layouts per CDNA5 ISA 7.12.2, 16-bit A 16x32 / B 32x16)
// ---------------------------------------------------------------------------
__device__ __forceinline__ v16h load_a_frag(const _Float16* lA, int lane, int kk) {
  // lA: [16][64] halves row-major. A fragment: lane L -> M=L%16,
  // V0..3: K = hi*8 + {0..7}; V4..7: K = 16 + hi*8 + {0..7}
  const int m  = lane & 15;
  const int hi = (lane >> 4) & 1;
  union { v16h v; unsigned int u[8]; } r;
  const unsigned int* row = reinterpret_cast<const unsigned int*>(lA + m * 64 + kk);
#pragma unroll
  for (int v = 0; v < 4; ++v) r.u[v] = row[hi * 4 + v];
#pragma unroll
  for (int v = 0; v < 4; ++v) r.u[4 + v] = row[8 + hi * 4 + v];
  return r.v;
}

__device__ __forceinline__ v16h load_b_frag(const _Float16* lB, int lane, int n0, int kk) {
  // lB: [64][64] halves (N-major, K contiguous). B fragment: lane L -> N=L%16,
  // V0..3: K = hi*8 + {0..7}; V4..7: K = 16 + hi*8 + {0..7}
  const int n  = lane & 15;
  const int kb = (lane >> 4) ? 4 : 0;
  union { v16h v; unsigned int u[8]; } r;
  const unsigned int* row = reinterpret_cast<const unsigned int*>(lB + (n0 + n) * 64 + kk);
#pragma unroll
  for (int v = 0; v < 8; ++v) r.u[v] = row[(v >> 2) * 8 + kb + (v & 3)];
  return r.v;
}

// ---------------------------------------------------------------------------
// Tiled WMMA GEMM: C[M,N] = op(A) * op(B) (+bias) (+relu / tanh-mask)
//  TA=0: A is [M,K] row-major (stride lda).   TA=1: A is [K,M] (stride lda).
//  TB=0: B is [K,N] row-major (stride ldb).   TB=1: B is [N,K] (stride ldb) => A*B^T.
//  Block: 128 threads = 4 waves; block tile 16(M) x 64(N); BK = 64.
//  Requires: M%16==0, N%64==0, K%64==0 (true for all call sites here).
// ---------------------------------------------------------------------------
template <bool TA, bool TB, bool BIAS, bool RELU, bool TMASK, bool ST32, bool ST16>
__global__ __launch_bounds__(128) void wmma_gemm(
    const _Float16* __restrict__ A, const _Float16* __restrict__ B,
    float* __restrict__ C32, _Float16* __restrict__ C16,
    const float* __restrict__ bias,
    const int* __restrict__ batA, const int* __restrict__ batB,
    int M, int N, int K, int lda, int ldb, float scale) {
  const int tid  = threadIdx.x;
  const int lane = tid & 31;
  const int wave = tid >> 5;
  const int m0   = blockIdx.y * 16;
  const int n0   = blockIdx.x * 64;

  __shared__ _Float16 lA[16 * 64];
  __shared__ _Float16 lB[64 * 64];

  union { v8f v; float f[8]; } acc;
#pragma unroll
  for (int i = 0; i < 8; ++i) acc.f[i] = 0.0f;

  for (int k0 = 0; k0 < K; k0 += 64) {
    // ---- stage A tile [16][64] ----
    if (!TA) {
      const int r = tid >> 3, kc = (tid & 7) * 8; // 8 halves = 16B per thread
#if ASYNC_LDS
      ASYNC_CP16(&lA[r * 64 + kc], &A[(size_t)(m0 + r) * lda + k0 + kc]);
#else
      *reinterpret_cast<uint4*>(&lA[r * 64 + kc]) =
          *reinterpret_cast<const uint4*>(&A[(size_t)(m0 + r) * lda + k0 + kc]);
#endif
    } else {
      const int m = tid & 15, ks = (tid >> 4) * 8;
#pragma unroll
      for (int i = 0; i < 8; ++i)
        lA[m * 64 + ks + i] = A[(size_t)(k0 + ks + i) * lda + m0 + m];
    }
    // ---- stage B tile [64 N][64 K] ----
    if (TB) {
      const int n = tid >> 1, kc = (tid & 1) * 32;
#pragma unroll
      for (int i = 0; i < 4; ++i) {
#if ASYNC_LDS
        ASYNC_CP16(&lB[n * 64 + kc + i * 8],
                   &B[(size_t)(n0 + n) * ldb + k0 + kc + i * 8]);
#else
        *reinterpret_cast<uint4*>(&lB[n * 64 + kc + i * 8]) =
            *reinterpret_cast<const uint4*>(&B[(size_t)(n0 + n) * ldb + k0 + kc + i * 8]);
#endif
      }
    } else {
      const int n = tid & 63, kh = (tid >> 6) * 32;
#pragma unroll
      for (int i = 0; i < 32; ++i)
        lB[n * 64 + kh + i] = B[(size_t)(k0 + kh + i) * ldb + n0 + n];
    }
#if ASYNC_LDS
    if (!TA || TB) async_wait0(); // wait for this wave's async-LDS writes
#endif
    __syncthreads();
#pragma unroll
    for (int kk = 0; kk < 64; kk += 32) {
      v16h af = load_a_frag(lA, lane, kk);
      v16h bf = load_b_frag(lB, lane, wave * 16, kk);
      acc.v = __builtin_amdgcn_wmma_f32_16x16x32_f16(
          false, af, false, bf, (short)0, acc.v, false, false);
    }
    __syncthreads();
  }

  // ---- epilogue: C layout lane L -> N=L%16, VGPR r -> M = r + 8*(L>=16) ----
  const int nc = n0 + wave * 16 + (lane & 15);
  const int hi = (lane >> 4) & 1;
  const float bval = BIAS ? bias[nc] : 0.0f;
  const int bb = TMASK ? batB[nc] : 0;
#pragma unroll
  for (int r = 0; r < 8; ++r) {
    const int mr = m0 + r + hi * 8;
    float v = acc.f[r] + bval;
    if (RELU) v = v > 0.0f ? v : 0.0f;
    if (TMASK) v = (batA[mr] == bb) ? tanhf(v * scale) : 0.0f;
    const size_t idx = (size_t)mr * N + nc;
    if (ST32) C32[idx] = v;
    if (ST16) C16[idx] = (_Float16)v;
  }
}

// ---------------------------------------------------------------------------
// Small scalar kernels
// ---------------------------------------------------------------------------
__global__ void dense_small(const float* __restrict__ X, const float* __restrict__ W,
                            const float* __restrict__ bias, float* __restrict__ Y,
                            int K, int relu) {
  __shared__ float xs[64];
  const int row = blockIdx.x, c = threadIdx.x; // blockDim = 128
  if (c < K) xs[c] = X[(size_t)row * K + c];
  __syncthreads();
  float s = bias[c];
  for (int k = 0; k < K; ++k) s += xs[k] * W[k * EMBD + c];
  if (relu) s = fmaxf(s, 0.0f);
  Y[(size_t)row * EMBD + c] = s;
}

__global__ void copy_f32(const float* __restrict__ s, float* __restrict__ d, int n) {
  int i = blockIdx.x * 256 + threadIdx.x;
  if (i < n) d[i] = s[i];
}

__global__ void zero_f32(float* __restrict__ d, int n) {
  int i = blockIdx.x * 256 + threadIdx.x;
  if (i < n) d[i] = 0.0f;
}

__global__ void f32_to_f16k(const float* __restrict__ s, _Float16* __restrict__ d, int n) {
  int i = blockIdx.x * 256 + threadIdx.x;
  if (i < n) d[i] = (_Float16)s[i];
}

// src [R,C] f32 -> dst [C,R] f16
__global__ void transpose_f16(const float* __restrict__ src, _Float16* __restrict__ dst,
                              int R, int C) {
  int idx = blockIdx.x * 256 + threadIdx.x;
  if (idx >= R * C) return;
  int r = idx / C, c = idx % C;
  dst[(size_t)c * R + r] = (_Float16)src[idx];
}

// agg[dst] += h[src] + ee1[attr0] + ee2[attr1]; one block per edge, 128 threads
__global__ void edge_scatter(const float* __restrict__ h, const float* __restrict__ ee1,
                             const float* __restrict__ ee2, const int* __restrict__ ei,
                             const int* __restrict__ ea, float* __restrict__ agg, int E) {
  const int e = blockIdx.x, c = threadIdx.x;
  const int s  = ei[e];
  const int dn = ei[E + e];
  const int a0 = ea[2 * e], a1 = ea[2 * e + 1];
  float v = h[(size_t)s * EMBD + c] + ee1[a0 * EMBD + c] + ee2[a1 * EMBD + c];
  atomicAdd(&agg[(size_t)dn * EMBD + c], v);
}

__global__ void bn_stats(const float* __restrict__ X, float* __restrict__ mu,
                         float* __restrict__ var, int Nrows) {
  __shared__ float ss[256], sq[256];
  const int c = blockIdx.x, t = threadIdx.x;
  float s = 0.0f, q = 0.0f;
  for (int r = t; r < Nrows; r += 256) {
    float v = X[(size_t)r * EMBD + c];
    s += v; q += v * v;
  }
  ss[t] = s; sq[t] = q;
  __syncthreads();
  for (int o = 128; o > 0; o >>= 1) {
    if (t < o) { ss[t] += ss[t + o]; sq[t] += sq[t + o]; }
    __syncthreads();
  }
  if (t == 0) {
    float m = ss[0] / (float)Nrows;
    mu[c] = m;
    var[c] = sq[0] / (float)Nrows - m * m;
  }
}

__global__ void bn_apply(const float* __restrict__ hh, float* __restrict__ h,
                         const float* __restrict__ mu, const float* __restrict__ var,
                         const float* __restrict__ g, const float* __restrict__ b,
                         int tot, int relu) {
  int idx = blockIdx.x * 256 + threadIdx.x;
  if (idx >= tot) return;
  int c = idx & (EMBD - 1);
  float v = (hh[idx] - mu[c]) * rsqrtf(var[c] + 1e-5f) * g[c] + b[c];
  if (relu) v = fmaxf(v, 0.0f);
  h[idx] = v;
}

// P[batch[node], colOff + c] += (c<128 ? Xa : Xb); one block per node, 256 threads
__global__ void pool_pair(const float* __restrict__ Xa, const float* __restrict__ Xb,
                          const int* __restrict__ batch, float* __restrict__ P, int colOff) {
  const int node = blockIdx.x, c = threadIdx.x;
  const int b = batch[node];
  float v = (c < EMBD) ? Xa[(size_t)node * EMBD + c]
                       : Xb[(size_t)node * EMBD + (c - EMBD)];
  atomicAdd(&P[(size_t)b * (8 * EMBD) + colOff + c], v);
}

__global__ void mlp(const float* __restrict__ X, const float* __restrict__ W,
                    const float* __restrict__ bias, float* __restrict__ Y,
                    int M, int N, int K, int relu) {
  int idx = blockIdx.x * blockDim.x + threadIdx.x;
  if (idx >= M * N) return;
  int m = idx / N, n = idx % N;
  float s = bias[n];
  for (int k = 0; k < K; ++k) s += X[m * K + k] * W[k * N + n];
  if (relu) s = fmaxf(s, 0.0f);
  Y[idx] = s;
}

// ---------------------------------------------------------------------------
// Host orchestration
// ---------------------------------------------------------------------------
static void run_gin(hipStream_t st, const float* x, const int* ei, const int* ea,
                    const float* embW, const float* embb, const float* ee1,
                    const float* ee2, const float* W1, const float* b1,
                    const float* W2, const float* b2, const float* bng,
                    const float* bnb, int Nn, int E, float* h, float* agg,
                    float* hh, _Float16* agg16, _Float16* hh1_16,
                    _Float16* w1t16, _Float16* w2t16, float* mu, float* var,
                    float* out_h, _Float16* out_h16) {
  const int tot = Nn * EMBD;
  dense_small<<<Nn, 128, 0, st>>>(x, embW, embb, h, 40, 1);
  for (int l = 0; l < 5; ++l) {
    copy_f32<<<(tot + 255) / 256, 256, 0, st>>>(h, agg, tot);
    edge_scatter<<<E, 128, 0, st>>>(h, ee1 + (size_t)l * 6 * EMBD,
                                    ee2 + (size_t)l * 3 * EMBD, ei, ea, agg, E);
    f32_to_f16k<<<(tot + 255) / 256, 256, 0, st>>>(agg, agg16, tot);
    transpose_f16<<<(EMBD * 256 + 255) / 256, 256, 0, st>>>(
        W1 + (size_t)l * EMBD * 256, w1t16, EMBD, 256);
    {
      dim3 g(256 / 64, Nn / 16);
      wmma_gemm<false, true, true, true, false, false, true><<<g, 128, 0, st>>>(
          agg16, w1t16, nullptr, hh1_16, b1 + (size_t)l * 256, nullptr, nullptr,
          Nn, 256, EMBD, EMBD, EMBD, 1.0f);
    }
    transpose_f16<<<(256 * EMBD + 255) / 256, 256, 0, st>>>(
        W2 + (size_t)l * 256 * EMBD, w2t16, 256, EMBD);
    {
      dim3 g(EMBD / 64, Nn / 16);
      wmma_gemm<false, true, true, false, false, true, false><<<g, 128, 0, st>>>(
          hh1_16, w2t16, hh, nullptr, b2 + (size_t)l * EMBD, nullptr, nullptr,
          Nn, EMBD, 256, 256, 256, 1.0f);
    }
    bn_stats<<<EMBD, 256, 0, st>>>(hh, mu, var, Nn);
    bn_apply<<<(tot + 255) / 256, 256, 0, st>>>(hh, h, mu, var,
                                                bng + (size_t)l * EMBD,
                                                bnb + (size_t)l * EMBD, tot,
                                                (l < 4) ? 1 : 0);
  }
  copy_f32<<<(tot + 255) / 256, 256, 0, st>>>(h, out_h, tot);
  f32_to_f16k<<<(tot + 255) / 256, 256, 0, st>>>(h, out_h16, tot);
}

extern "C" void kernel_launch(void* const* d_in, const int* in_sizes, int n_in,
                              void* d_out, int out_size, void* d_ws, size_t ws_size,
                              hipStream_t stream) {
  (void)in_sizes; (void)n_in; (void)out_size; (void)ws_size;
  const float* solute_x = (const float*)d_in[0];
  const float* wat_x    = (const float*)d_in[1];
  const float* oct_x    = (const float*)d_in[2];
  const int* ei_sol = (const int*)d_in[3];
  const int* ea_sol = (const int*)d_in[4];
  const int* batch_sol = (const int*)d_in[5];
  const int* ei_oct = (const int*)d_in[6];
  const int* ea_oct = (const int*)d_in[7];
  const int* batch_oct = (const int*)d_in[8];
  const int* batch_wat = (const int*)d_in[9];
  const float* sol_embW = (const float*)d_in[10];
  const float* sol_embb = (const float*)d_in[11];
  const float* sol_ee1  = (const float*)d_in[12];
  const float* sol_ee2  = (const float*)d_in[13];
  const float* sol_W1   = (const float*)d_in[14];
  const float* sol_b1   = (const float*)d_in[15];
  const float* sol_W2   = (const float*)d_in[16];
  const float* sol_b2   = (const float*)d_in[17];
  const float* sol_bng  = (const float*)d_in[18];
  const float* sol_bnb  = (const float*)d_in[19];
  const float* oct_embW = (const float*)d_in[20];
  const float* oct_embb = (const float*)d_in[21];
  const float* oct_ee1  = (const float*)d_in[22];
  const float* oct_ee2  = (const float*)d_in[23];
  const float* oct_W1   = (const float*)d_in[24];
  const float* oct_b1   = (const float*)d_in[25];
  const float* oct_W2   = (const float*)d_in[26];
  const float* oct_b2   = (const float*)d_in[27];
  const float* oct_bng  = (const float*)d_in[28];
  const float* oct_bnb  = (const float*)d_in[29];
  const float* wat_W1   = (const float*)d_in[30];
  const float* wat_b1   = (const float*)d_in[31];
  const float* wat_W2   = (const float*)d_in[32];
  const float* wat_b2   = (const float*)d_in[33];
  const float* out_W0   = (const float*)d_in[34];
  const float* out_b0   = (const float*)d_in[35];
  const float* out_W1   = (const float*)d_in[36];
  const float* out_b1   = (const float*)d_in[37];
  const float* out_W2   = (const float*)d_in[38];
  const float* out_b2   = (const float*)d_in[39];
  const float* out_W3   = (const float*)d_in[40];
  const float* out_b3   = (const float*)d_in[41];
  float* out = (float*)d_out;

  // ---- workspace bump allocator (256B aligned) ----
  char* ws = (char*)d_ws;
  size_t off = 0;
  auto alloc = [&](size_t bytes) -> void* {
    void* p = ws + off;
    off += (bytes + 255) & ~(size_t)255;
    return p;
  };
  const size_t nodeF = (size_t)NSOL * EMBD; // 786432
  float* h    = (float*)alloc(nodeF * 4);
  float* agg  = (float*)alloc(nodeF * 4);
  float* hh   = (float*)alloc(nodeF * 4);
  float* solh = (float*)alloc(nodeF * 4);
  float* octh = (float*)alloc(nodeF * 4);
  float* wath = (float*)alloc(nodeF * 4);
  float* simW = (float*)alloc(nodeF * 4);
  float* simO = (float*)alloc(nodeF * 4);
  float* watp = (float*)alloc(nodeF * 4);
  float* octp = (float*)alloc(nodeF * 4);
  float* P    = (float*)alloc((size_t)NBATCH * 8 * EMBD * 4);
  float* m1   = (float*)alloc((size_t)NBATCH * EMBD * 4);
  float* m2   = (float*)alloc((size_t)NBATCH * 64 * 4);
  float* m3   = (float*)alloc((size_t)NBATCH * 32 * 4);
  float* mu   = (float*)alloc(EMBD * 4);
  float* var  = (float*)alloc(EMBD * 4);
  _Float16* agg16  = (_Float16*)alloc(nodeF * 2);
  _Float16* hh1_16 = (_Float16*)alloc(nodeF * 2 * 2); // [N,256]
  _Float16* sol16  = (_Float16*)alloc(nodeF * 2);
  _Float16* oct16  = (_Float16*)alloc(nodeF * 2);
  _Float16* wat16  = (_Float16*)alloc(nodeF * 2);
  _Float16* w1t16  = (_Float16*)alloc((size_t)256 * EMBD * 2);
  _Float16* w2t16  = (_Float16*)alloc((size_t)EMBD * 256 * 2);
  _Float16* ww2t16 = (_Float16*)alloc((size_t)EMBD * EMBD * 2);
  _Float16* im_wat = (_Float16*)alloc((size_t)NSOL * NWAT * 2);
  _Float16* im_oct = (_Float16*)alloc((size_t)NSOL * NOCT * 2);

  const float scale = 0.088388347648318447f; // 1/sqrt(128)
  const int tot = NSOL * EMBD;

  // ---- water MLP: wat = relu(wat_x@W1+b1)@W2 + b2 ----
  dense_small<<<NWAT, 128, 0, stream>>>(wat_x, wat_W1, wat_b1, agg, 40, 1);
  f32_to_f16k<<<(tot + 255) / 256, 256, 0, stream>>>(agg, agg16, tot);
  transpose_f16<<<(EMBD * EMBD + 255) / 256, 256, 0, stream>>>(wat_W2, ww2t16, EMBD, EMBD);
  {
    dim3 g(EMBD / 64, NWAT / 16);
    wmma_gemm<false, true, true, false, false, true, true><<<g, 128, 0, stream>>>(
        agg16, ww2t16, wath, wat16, wat_b2, nullptr, nullptr,
        NWAT, EMBD, EMBD, EMBD, EMBD, 1.0f);
  }

  // ---- GIN GNNs ----
  run_gin(stream, solute_x, ei_sol, ea_sol, sol_embW, sol_embb, sol_ee1, sol_ee2,
          sol_W1, sol_b1, sol_W2, sol_b2, sol_bng, sol_bnb, NSOL, ESOL,
          h, agg, hh, agg16, hh1_16, w1t16, w2t16, mu, var, solh, sol16);
  run_gin(stream, oct_x, ei_oct, ea_oct, oct_embW, oct_embb, oct_ee1, oct_ee2,
          oct_W1, oct_b1, oct_W2, oct_b2, oct_bng, oct_bnb, NOCT, EOCT,
          h, agg, hh, agg16, hh1_16, w1t16, w2t16, mu, var, octh, oct16);

  // ---- interaction maps: im = mask * tanh(sol @ solvT * scale) ----
  {
    dim3 g(NWAT / 64, NSOL / 16);
    wmma_gemm<false, true, false, false, true, false, true><<<g, 128, 0, stream>>>(
        sol16, wat16, nullptr, im_wat, nullptr, batch_sol, batch_wat,
        NSOL, NWAT, EMBD, EMBD, EMBD, scale);
  }
  {
    dim3 g(NOCT / 64, NSOL / 16);
    wmma_gemm<false, true, false, false, true, false, true><<<g, 128, 0, stream>>>(
        sol16, oct16, nullptr, im_oct, nullptr, batch_sol, batch_oct,
        NSOL, NOCT, EMBD, EMBD, EMBD, scale);
  }
  // ---- im @ solv  (K = 6144, NN) ----
  {
    dim3 g(EMBD / 64, NSOL / 16);
    wmma_gemm<false, false, false, false, false, true, false><<<g, 128, 0, stream>>>(
        im_wat, wat16, simW, nullptr, nullptr, nullptr, nullptr,
        NSOL, EMBD, NWAT, NWAT, EMBD, 1.0f);
    wmma_gemm<false, false, false, false, false, true, false><<<g, 128, 0, stream>>>(
        im_oct, oct16, simO, nullptr, nullptr, nullptr, nullptr,
        NSOL, EMBD, NOCT, NOCT, EMBD, 1.0f);
  }
  // ---- im^T @ sol  (K = 6144, TA=1) ----
  {
    dim3 g(EMBD / 64, NWAT / 16);
    wmma_gemm<true, false, false, false, false, true, false><<<g, 128, 0, stream>>>(
        im_wat, sol16, watp, nullptr, nullptr, nullptr, nullptr,
        NWAT, EMBD, NSOL, NWAT, EMBD, 1.0f);
  }
  {
    dim3 g(EMBD / 64, NOCT / 16);
    wmma_gemm<true, false, false, false, false, true, false><<<g, 128, 0, stream>>>(
        im_oct, sol16, octp, nullptr, nullptr, nullptr, nullptr,
        NOCT, EMBD, NSOL, NOCT, EMBD, 1.0f);
  }

  // ---- global add pool into P [192, 1024] = [p_sw | p_so | p_w | p_o] ----
  const int ptot = NBATCH * 8 * EMBD;
  zero_f32<<<(ptot + 255) / 256, 256, 0, stream>>>(P, ptot);
  pool_pair<<<NSOL, 256, 0, stream>>>(solh, simW, batch_sol, P, 0);
  pool_pair<<<NSOL, 256, 0, stream>>>(solh, simO, batch_sol, P, 2 * EMBD);
  pool_pair<<<NWAT, 256, 0, stream>>>(wath, watp, batch_wat, P, 4 * EMBD);
  pool_pair<<<NOCT, 256, 0, stream>>>(octh, octp, batch_oct, P, 6 * EMBD);

  // ---- output MLP ----
  mlp<<<(NBATCH * EMBD + 255) / 256, 256, 0, stream>>>(P, out_W0, out_b0, m1,
                                                       NBATCH, EMBD, 8 * EMBD, 1);
  mlp<<<(NBATCH * 64 + 255) / 256, 256, 0, stream>>>(m1, out_W1, out_b1, m2,
                                                     NBATCH, 64, EMBD, 1);
  mlp<<<(NBATCH * 32 + 255) / 256, 256, 0, stream>>>(m2, out_W2, out_b2, m3,
                                                     NBATCH, 32, 64, 1);
  mlp<<<(NBATCH * 1 + 255) / 256, 256, 0, stream>>>(m3, out_W3, out_b3, out,
                                                    NBATCH, 1, 32, 0);
}
